// minerva_56152402427977
// MI455X (gfx1250) — compile-verified, matrix operand order checked by reading, not run on hardware
//
#include <hip/hip_runtime.h>

// ---------------------------------------------------------------------------
// Sizes (compile-time, match the reference)
// ---------------------------------------------------------------------------
#define B_ROWS   4096
#define D_IN     512
#define D_FEAT   512
#define N_EX     20000
#define NEX_PAD  20096      // 157 * 128
#define N_LAB    100
#define LAB_PAD  112        // 7 * 16
#define NCHUNK   157        // NEX_PAD / 128
#define EPSF     1e-12f

typedef __attribute__((ext_vector_type(16))) _Float16 v16h;
typedef __attribute__((ext_vector_type(8)))  _Float16 v8h;
typedef __attribute__((ext_vector_type(8)))  float    v8f;

__device__ __forceinline__ v8f wmma_f16(v16h a, v16h b, v8f c) {
  // D = A(16x32 f16) * B(32x16 f16) + C(16x16 f32)
  return __builtin_amdgcn_wmma_f32_16x16x32_f16(
      /*neg_a=*/false, a, /*neg_b=*/false, b,
      /*c_mod=*/(short)0, c, /*reuse_a=*/false, /*reuse_b=*/false);
}

__device__ __forceinline__ v16h zero16h() {
  v16h r;
#pragma unroll
  for (int i = 0; i < 16; ++i) r[i] = (_Float16)0.f;
  return r;
}

// A/B fragment (16-bit, 16x32 / 32x16): lane holds K = {kb..kb+7, kb+16..kb+23}
// from a row-major row pointer; f32 source, converted to f16.
__device__ __forceinline__ v16h pack16_f32(const float* __restrict__ row, int kb) {
  float4 a0 = *(const float4*)(row + kb);
  float4 a1 = *(const float4*)(row + kb + 4);
  float4 b0 = *(const float4*)(row + kb + 16);
  float4 b1 = *(const float4*)(row + kb + 20);
  v16h r;
  r[0]  = (_Float16)a0.x; r[1]  = (_Float16)a0.y; r[2]  = (_Float16)a0.z; r[3]  = (_Float16)a0.w;
  r[4]  = (_Float16)a1.x; r[5]  = (_Float16)a1.y; r[6]  = (_Float16)a1.z; r[7]  = (_Float16)a1.w;
  r[8]  = (_Float16)b0.x; r[9]  = (_Float16)b0.y; r[10] = (_Float16)b0.z; r[11] = (_Float16)b0.w;
  r[12] = (_Float16)b1.x; r[13] = (_Float16)b1.y; r[14] = (_Float16)b1.z; r[15] = (_Float16)b1.w;
  return r;
}

// Same fragment pattern from an f16 row pointer (two b128 loads).
__device__ __forceinline__ v16h pack16_f16(const _Float16* __restrict__ row, int kb) {
  v8h lo = *(const v8h*)(row + kb);
  v8h hi = *(const v8h*)(row + kb + 16);
  v16h r;
#pragma unroll
  for (int i = 0; i < 8; ++i) { r[i] = lo[i]; r[i + 8] = hi[i]; }
  return r;
}

// ---------------------------------------------------------------------------
// Kernel 1: Y[m,n] = l2norm_row( X[m,:] @ W[n,:] ), written as f16.
// Block = 16 rows x 512 cols, 8 waves, wave w owns cols [w*64, w*64+64).
// ---------------------------------------------------------------------------
__global__ __launch_bounds__(256) void proj_norm_kernel(
    const float* __restrict__ X, const float* __restrict__ W,
    _Float16* __restrict__ Y, int m_real) {
  const int tid  = threadIdx.x;
  const int wv   = tid >> 5;
  const int lane = tid & 31;
  const int ln   = lane & 15;
  const int hb   = (lane >> 4) & 1;
  const int rowbase = blockIdx.x * 16;

  __shared__ float ssq[16];
  if (tid < 16) ssq[tid] = 0.f;
  __syncthreads();

  const int  rA  = rowbase + ln;
  const bool okA = (rA < m_real);
  const float* arow = X + (size_t)rA * D_IN;

  v8f acc[4];
#pragma unroll
  for (int t = 0; t < 4; ++t)
#pragma unroll
    for (int i = 0; i < 8; ++i) acc[t][i] = 0.f;

  for (int kk = 0; kk < D_IN / 32; ++kk) {
    const int kb = kk * 32 + hb * 8;
    v16h a = zero16h();
    if (okA) a = pack16_f32(arow, kb);
#pragma unroll
    for (int t = 0; t < 4; ++t) {
      const int col = wv * 64 + t * 16 + ln;
      v16h b = pack16_f32(W + (size_t)col * D_IN, kb);
      acc[t] = wmma_f16(a, b, acc[t]);
    }
  }

  // Row sum-of-squares across the full 512 columns (lane-shuffle then LDS).
  float pr[8];
#pragma unroll
  for (int r = 0; r < 8; ++r) {
    float v = 0.f;
#pragma unroll
    for (int t = 0; t < 4; ++t) { float x = acc[t][r]; v += x * x; }
    pr[r] = v;
  }
#pragma unroll
  for (int msk = 1; msk < 16; msk <<= 1)
#pragma unroll
    for (int r = 0; r < 8; ++r) pr[r] += __shfl_xor(pr[r], msk, 32);
  if (ln == 0) {
#pragma unroll
    for (int r = 0; r < 8; ++r) atomicAdd(&ssq[r + hb * 8], pr[r]);
  }
  __syncthreads();

#pragma unroll
  for (int r = 0; r < 8; ++r) {
    const float inv = 1.f / fmaxf(sqrtf(ssq[r + hb * 8]), EPSF);
    const int   row = rowbase + r + hb * 8;
#pragma unroll
    for (int t = 0; t < 4; ++t) {
      const int col = wv * 64 + t * 16 + ln;
      Y[(size_t)row * D_FEAT + col] = (_Float16)(acc[t][r] * inv);
    }
  }
}

// ---------------------------------------------------------------------------
// Kernel 2: ecrT[lab, j] = (|x| row-L1-normalized ex_classes)[j, lab],
// zero-padded to LAB_PAD x NEX_PAD, stored transposed for B-fragment loads.
// One wave per exemplar row.
// ---------------------------------------------------------------------------
__global__ __launch_bounds__(256) void ecr_kernel(
    const float* __restrict__ exc, _Float16* __restrict__ ecrT) {
  const int row  = blockIdx.x * 8 + (threadIdx.x >> 5);
  const int lane = threadIdx.x & 31;
  const bool live = (row < N_EX);

  float v[4];
  float s = 0.f;
#pragma unroll
  for (int i = 0; i < 4; ++i) {
    const int c = lane + i * 32;
    float x = (live && c < N_LAB) ? exc[(size_t)row * N_LAB + c] : 0.f;
    v[i] = x;
    s += fabsf(x);
  }
#pragma unroll
  for (int msk = 16; msk; msk >>= 1) s += __shfl_xor(s, msk, 32);
  const float inv = 1.f / fmaxf(s, EPSF);
#pragma unroll
  for (int i = 0; i < 4; ++i) {
    const int c = lane + i * 32;
    if (c < LAB_PAD)
      ecrT[(size_t)c * NEX_PAD + row] = (_Float16)((c < N_LAB) ? v[i] * inv : 0.f);
  }
}

// ---------------------------------------------------------------------------
// Kernel 3: zero the global partial accumulators (ws is poisoned once only).
// ---------------------------------------------------------------------------
__global__ void zero_kernel(float* __restrict__ p, int n) {
  const int i = blockIdx.x * 256 + threadIdx.x;
  if (i < n) p[i] = 0.f;
}

// ---------------------------------------------------------------------------
// Kernel 4 (fused): for a 64-row tile of f_norm and a range of 128-exemplar
// chunks: S = Fn @ Enᵀ (WMMA), a = S^3 in registers, accumulate row L1 |a|,
// route a (f16) through LDS to A-layout, echoAcc += a @ ecr (WMMA).
// Partials combined with global f32 atomics.
// grid.x = 64 row-blocks, grid.y = 8 exemplar splits, 256 threads.
// ---------------------------------------------------------------------------
__global__ __launch_bounds__(256) void fused_kernel(
    const _Float16* __restrict__ Fn, const _Float16* __restrict__ En,
    const _Float16* __restrict__ EcrT,
    float* __restrict__ gAcc, float* __restrict__ gRs) {
  __shared__ _Float16 sA[64 * 136];   // cubed similarities, pad stride 136
  __shared__ float    rs[64];         // per-row L1 partials for this block

  const int tid  = threadIdx.x;
  const int wv   = tid >> 5;
  const int lane = tid & 31;
  const int ln   = lane & 15;
  const int hb   = (lane >> 4) & 1;
  const int mb   = blockIdx.x;
  const int m    = wv & 3;            // M-tile (0..3) within the 64-row block
  const int ngrp = wv >> 2;           // 0: n-tiles 0..3, 1: n-tiles 4..7(/4..6)

  if (tid < 64) rs[tid] = 0.f;

  v8f eAcc[4];
#pragma unroll
  for (int t = 0; t < 4; ++t)
#pragma unroll
    for (int i = 0; i < 8; ++i) eAcc[t][i] = 0.f;

  const _Float16* frow = Fn + (size_t)(mb * 64 + m * 16 + ln) * D_FEAT;
  __syncthreads();

  const int c0 = blockIdx.y * 20;
  const int c1 = min(NCHUNK, c0 + 20);

  for (int ch = c0; ch < c1; ++ch) {
    const int jbase = ch * 128;
    if (ch + 1 < c1)  // warm the next exemplar chunk (global_prefetch)
      __builtin_prefetch(En + (size_t)(jbase + 128 + (tid & 127)) * D_FEAT, 0, 1);

    // ---- similarity GEMM: 16x16 tiles, K = 512 -------------------------
    v8f s[4];
#pragma unroll
    for (int t = 0; t < 4; ++t)
#pragma unroll
      for (int i = 0; i < 8; ++i) s[t][i] = 0.f;

    for (int kk = 0; kk < D_FEAT / 32; ++kk) {
      const int kb = kk * 32 + hb * 8;
      v16h a = pack16_f16(frow, kb);
#pragma unroll
      for (int t = 0; t < 4; ++t) {
        const int j = jbase + ngrp * 64 + t * 16 + ln;
        v16h b = pack16_f16(En + (size_t)j * D_FEAT, kb);
        s[t] = wmma_f16(a, b, s[t]);
      }
    }

    // ---- a = s^3 (== sign(s)|s|^3), row |a| sums, stage into LDS -------
    float pr[8];
#pragma unroll
    for (int r = 0; r < 8; ++r) pr[r] = 0.f;
#pragma unroll
    for (int t = 0; t < 4; ++t) {
      const int colb = ngrp * 64 + t * 16 + ln;
#pragma unroll
      for (int r = 0; r < 8; ++r) {
        const float x  = s[t][r];
        const float a3 = x * x * x;
        pr[r] += fabsf(a3);
        sA[(m * 16 + r + hb * 8) * 136 + colb] = (_Float16)a3;
      }
    }
#pragma unroll
    for (int msk = 1; msk < 16; msk <<= 1)
#pragma unroll
      for (int r = 0; r < 8; ++r) pr[r] += __shfl_xor(pr[r], msk, 32);
    if (ln == 0) {
#pragma unroll
      for (int r = 0; r < 8; ++r) atomicAdd(&rs[m * 16 + r + hb * 8], pr[r]);
    }
    __syncthreads();

    // ---- echo GEMM: [64 x 128] (f16, from LDS) @ ecrT-chunk [128 x 112] -
    {
      const _Float16* arow = sA + (m * 16 + ln) * 136;
      const int nt = (ngrp == 0) ? 4 : 3;  // 7 N-tiles split 4 + 3
      for (int kk = 0; kk < 4; ++kk) {
        const int kb = kk * 32 + hb * 8;
        v16h a = pack16_f16(arow, kb);
        for (int t = 0; t < nt; ++t) {
          const int labt = ngrp * 4 + t;
          const _Float16* brow = EcrT + (size_t)(labt * 16 + ln) * NEX_PAD + jbase;
          v16h b = pack16_f16(brow, kb);
          eAcc[t] = wmma_f16(a, b, eAcc[t]);
        }
      }
    }
    __syncthreads();
  }

  // ---- combine partials across exemplar splits ------------------------
  {
    const int nt = (ngrp == 0) ? 4 : 3;
    for (int t = 0; t < nt; ++t) {
      const int lab = (ngrp * 4 + t) * 16 + ln;
#pragma unroll
      for (int r = 0; r < 8; ++r) {
        const int row = mb * 64 + m * 16 + r + hb * 8;
        atomicAdd(&gAcc[(size_t)row * LAB_PAD + lab], eAcc[t][r]);
      }
    }
  }
  if (tid < 64) atomicAdd(&gRs[mb * 64 + tid], rs[tid]);
}

// ---------------------------------------------------------------------------
// Kernel 5: echo = acc / max(rowL1, eps), drop label padding.
// ---------------------------------------------------------------------------
__global__ void finalize_kernel(const float* __restrict__ gAcc,
                                const float* __restrict__ gRs,
                                float* __restrict__ out) {
  const int i = blockIdx.x * 256 + threadIdx.x;
  if (i >= B_ROWS * N_LAB) return;
  const int row = i / N_LAB;
  const int lab = i - row * N_LAB;
  out[i] = gAcc[(size_t)row * LAB_PAD + lab] / fmaxf(gRs[row], EPSF);
}

// ---------------------------------------------------------------------------
// Workspace layout (bytes): ~29.7 MB total
// ---------------------------------------------------------------------------
#define OFF_FN   ((size_t)0)                                   // f16 [4096][512]
#define OFF_EN   (OFF_FN + (size_t)B_ROWS * D_FEAT * 2)        // f16 [20096][512]
#define OFF_ECRT (OFF_EN + (size_t)NEX_PAD * D_FEAT * 2)       // f16 [112][20096]
#define OFF_ACC  (OFF_ECRT + (size_t)LAB_PAD * NEX_PAD * 2)    // f32 [4096][112]
#define OFF_RS   (OFF_ACC + (size_t)B_ROWS * LAB_PAD * 4)      // f32 [4096]

extern "C" void kernel_launch(void* const* d_in, const int* in_sizes, int n_in,
                              void* d_out, int out_size, void* d_ws, size_t ws_size,
                              hipStream_t stream) {
  (void)in_sizes; (void)n_in; (void)out_size; (void)ws_size;
  const float* features    = (const float*)d_in[0];
  const float* g_weight    = (const float*)d_in[1];
  const float* ex_features = (const float*)d_in[2];
  const float* ex_classes  = (const float*)d_in[3];
  // d_in[4] class_reps is the identity (class_dim is None) -> folded away.

  char* ws = (char*)d_ws;
  _Float16* fn   = (_Float16*)(ws + OFF_FN);
  _Float16* en   = (_Float16*)(ws + OFF_EN);
  _Float16* ecrT = (_Float16*)(ws + OFF_ECRT);
  float*    gAcc = (float*)(ws + OFF_ACC);
  float*    gRs  = (float*)(ws + OFF_RS);
  float*    out  = (float*)d_out;

  const int nzero = B_ROWS * LAB_PAD + B_ROWS;  // gAcc ++ gRs (contiguous)
  zero_kernel<<<(nzero + 255) / 256, 256, 0, stream>>>(gAcc, nzero);

  proj_norm_kernel<<<B_ROWS / 16, 256, 0, stream>>>(features, g_weight, fn, B_ROWS);
  proj_norm_kernel<<<NEX_PAD / 16, 256, 0, stream>>>(ex_features, g_weight, en, N_EX);
  ecr_kernel<<<NEX_PAD / 8, 256, 0, stream>>>(ex_classes, ecrT);

  dim3 fg(B_ROWS / 64, 8);
  fused_kernel<<<fg, 256, 0, stream>>>(fn, en, ecrT, gAcc, gRs);

  finalize_kernel<<<(B_ROWS * N_LAB + 255) / 256, 256, 0, stream>>>(gAcc, gRs, out);
}
